// NetGlobGATFix_69569880261289
// MI455X (gfx1250) — matrix operand order repaired and checked by reading, not compiled, not run on hardware
//
#include <hip/hip_runtime.h>
#include <hip/hip_bf16.h>
#include <cmath>
#include <stdint.h>

#define N_NODES 10000
#define N_EDGES 160000
#define E_TOT   (N_EDGES + N_NODES)

typedef __attribute__((ext_vector_type(16))) _Float16 v16h;
typedef __attribute__((ext_vector_type(8)))  _Float16 v8h;
typedef __attribute__((ext_vector_type(4)))  _Float16 v4h;
typedef __attribute__((ext_vector_type(8)))  float    v8f;

union FragAB { v16h v; v8h half8[2]; _Float16 h[16]; };
union FragC  { v8f  v; float f[8]; };

__device__ __forceinline__ float selu_f(float v) {
  const float kScale = 1.0507009873554805f;
  const float kAlpha = 1.6732632423543772f;
  return v > 0.f ? kScale * v : kScale * kAlpha * (expf(v) - 1.f);
}

__device__ __forceinline__ void edge_sd(const long long* ei, int e, int& s, int& d) {
  if (e < N_EDGES) { s = (int)ei[e]; d = (int)ei[N_EDGES + e]; }
  else             { s = e - N_EDGES; d = s; }
}

__device__ __forceinline__ void atomicMaxF(float* addr, float val) {
  if (val >= 0.f) atomicMax((int*)addr, __float_as_int(val));
  else            atomicMin((unsigned int*)addr, (unsigned int)__float_as_int(val));
}

// ---------------------------------------------------------------- CNN ------
__global__ void conv3x3_selu_kernel(const float* __restrict__ in,
                                    const float* __restrict__ w,
                                    const float* __restrict__ bias,
                                    float* __restrict__ out, int Cin, int Cout) {
  int pix = blockIdx.x * blockDim.x + threadIdx.x;   // 4096 pixels
  int co  = blockIdx.y;
  if (pix >= 4096 || co >= Cout) return;
  int py = pix >> 6, px = pix & 63;
  float acc = bias[co];
  for (int ci = 0; ci < Cin; ++ci) {
    const float* ip = in + (size_t)ci * 4096;
    const float* wp = w + ((size_t)co * Cin + ci) * 9;
    for (int ky = 0; ky < 3; ++ky) {
      int yy = py + ky - 1;
      if (yy < 0 || yy > 63) continue;
      for (int kx = 0; kx < 3; ++kx) {
        int xx = px + kx - 1;
        if (xx < 0 || xx > 63) continue;
        acc += ip[yy * 64 + xx] * wp[ky * 3 + kx];
      }
    }
  }
  out[(size_t)co * 4096 + pix] = selu_f(acc);
}

__global__ void avgpool_kernel(const float* __restrict__ in, float* __restrict__ gfeat) {
  __shared__ float sm[256];
  int c = blockIdx.x, t = threadIdx.x;
  float s = 0.f;
  for (int i = t; i < 4096; i += 256) s += in[(size_t)c * 4096 + i];
  sm[t] = s;
  __syncthreads();
  for (int k = 128; k > 0; k >>= 1) {
    if (t < k) sm[t] += sm[t + k];
    __syncthreads();
  }
  if (t == 0) gfeat[c] = sm[0] / 4096.f;
}

__global__ void build_h0_kernel(const float* __restrict__ x,
                                const float* __restrict__ gfeat,
                                float* __restrict__ h0) {
  int idx = blockIdx.x * blockDim.x + threadIdx.x;
  if (idx >= N_NODES * 34) return;
  int n = idx / 34, f = idx - n * 34;
  h0[idx] = (f < 24) ? gfeat[f] : x[n * 10 + (f - 24)];
}

// ------------------------------------------------------ WMMA f16 GEMM -----
// C(M x N) = A(M x K) @ B(K x N); f32 in/out, v_wmma_f32_16x16x32_f16.
// 128 threads = 4 waves; block tile 64x64; wave = 16(M) x 64(N) strip.
// Fast path: double-buffered LDS + register prefetch (global loads for tile
// k+1 issued before the 4 WMMAs of tile k; convert+ds_store after).
#define BM 64
#define BN 64
#define BK 32
#define LDP 48   // LDS pitch in halves (96 B = multiple of 16 B)

__device__ __forceinline__ void gemm_load_tiles(float4 ra[4], float rb[4][4],
    const float* __restrict__ A, const float* __restrict__ B,
    int bm, int bn, int k0, int K, int N, int tid) {
#pragma unroll
  for (int i = 0; i < 4; ++i) {            // A: 64x32, float4 along k
    int chunk = tid + i * 128;             // 512 chunks
    int r  = chunk >> 3;                   // 8 chunks per 32-wide row
    int c4 = (chunk & 7) * 4;
    ra[i] = *(const float4*)&A[(size_t)(bm + r) * K + (k0 + c4)];
  }
#pragma unroll
  for (int i = 0; i < 4; ++i) {            // B: 32x64, 4 scalars along k (coalesced in n)
    int chunk = tid + i * 128;             // 512 chunks
    int n  = chunk & 63;
    int r4 = (chunk >> 6) * 4;
#pragma unroll
    for (int j = 0; j < 4; ++j)
      rb[i][j] = B[(size_t)(k0 + r4 + j) * N + (bn + n)];
  }
}

__device__ __forceinline__ void gemm_store_tiles(const float4 ra[4], const float rb[4][4],
    _Float16 (*__restrict__ Asb)[LDP], _Float16 (*__restrict__ Btb)[LDP], int tid) {
#pragma unroll
  for (int i = 0; i < 4; ++i) {
    int chunk = tid + i * 128;
    int r  = chunk >> 3;
    int c4 = (chunk & 7) * 4;
    v4h hv;
    hv.x = (_Float16)ra[i].x; hv.y = (_Float16)ra[i].y;
    hv.z = (_Float16)ra[i].z; hv.w = (_Float16)ra[i].w;
    *(v4h*)&Asb[r][c4] = hv;               // ds_store_b64
  }
#pragma unroll
  for (int i = 0; i < 4; ++i) {
    int chunk = tid + i * 128;
    int n  = chunk & 63;
    int r4 = (chunk >> 6) * 4;
    v4h hv;
    hv.x = (_Float16)rb[i][0]; hv.y = (_Float16)rb[i][1];
    hv.z = (_Float16)rb[i][2]; hv.w = (_Float16)rb[i][3];
    *(v4h*)&Btb[n][r4] = hv;               // ds_store_b64 (transposed [n][k])
  }
}

__device__ __forceinline__ void gemm_compute(const _Float16 (*__restrict__ Asb)[LDP],
                                             const _Float16 (*__restrict__ Btb)[LDP],
                                             FragC acc[4], int wave, int l15, int khigh) {
  // A fragment (ISA 7.12.2): halves 0-7 = row[khigh*8..+7], 8-15 = row[16+khigh*8..+7]
  FragAB fa;
  int am = wave * 16 + l15;
  fa.half8[0] = *(const v8h*)&Asb[am][khigh * 8];
  fa.half8[1] = *(const v8h*)&Asb[am][16 + khigh * 8];
#pragma unroll
  for (int nt = 0; nt < 4; ++nt) {
    // B fragment: halves 0-7 = Bt[n][khigh*8..+7], 8-15 = Bt[n][16+khigh*8..+7]
    FragAB fb;
    int nc = nt * 16 + l15;
    fb.half8[0] = *(const v8h*)&Btb[nc][khigh * 8];
    fb.half8[1] = *(const v8h*)&Btb[nc][16 + khigh * 8];
    acc[nt].v = __builtin_amdgcn_wmma_f32_16x16x32_f16(
        false, fa.v, false, fb.v, (short)0, acc[nt].v, false, false);
  }
}

__global__ __launch_bounds__(128)
void gemm_wmma_kernel(const float* __restrict__ A, const float* __restrict__ B,
                      float* __restrict__ Cmat, int M, int K, int N) {
  __shared__ __align__(16) _Float16 As[2][BM][LDP];  // [buf][m][k]
  __shared__ __align__(16) _Float16 Bt[2][BN][LDP];  // [buf][n][k]

  int tid   = threadIdx.x;
  int lane  = tid & 31;
  int wave  = tid >> 5;
  int bm    = blockIdx.y * BM;
  int bn    = blockIdx.x * BN;
  int l15   = lane & 15;
  int khigh = lane >> 4;

  FragC acc[4];
#pragma unroll
  for (int nt = 0; nt < 4; ++nt)
#pragma unroll
    for (int q = 0; q < 8; ++q) acc[nt].f[q] = 0.f;

  bool fast = (bm + BM <= M) && (bn + BN <= N) && ((K & (BK - 1)) == 0);

  if (fast) {
    float4 ra[4];
    float  rb[4][4];
    gemm_load_tiles(ra, rb, A, B, bm, bn, 0, K, N, tid);
    gemm_store_tiles(ra, rb, As[0], Bt[0], tid);
    int p = 0;
    for (int k0 = 0; k0 < K; k0 += BK) {
      __syncthreads();
      bool has_next = (k0 + BK < K);
      if (has_next) gemm_load_tiles(ra, rb, A, B, bm, bn, k0 + BK, K, N, tid);
      gemm_compute(As[p], Bt[p], acc, wave, l15, khigh);       // hides load latency
      if (has_next) gemm_store_tiles(ra, rb, As[1 - p], Bt[1 - p], tid);
      p ^= 1;
    }
  } else {
    for (int k0 = 0; k0 < K; k0 += BK) {
      for (int i = tid; i < BM * BK; i += 128) {
        int r = i >> 5, c = i & 31;
        int gr = bm + r, gc = k0 + c;
        As[0][r][c] = (_Float16)((gr < M && gc < K) ? A[(size_t)gr * K + gc] : 0.f);
      }
      for (int i = tid; i < BK * BN; i += 128) {
        int r = i >> 6, c = i & 63;         // r = k, c = n
        int gr = k0 + r, gc = bn + c;
        Bt[0][c][r] = (_Float16)((gr < K && gc < N) ? B[(size_t)gr * N + gc] : 0.f);
      }
      __syncthreads();
      gemm_compute(As[0], Bt[0], acc, wave, l15, khigh);
      __syncthreads();
    }
  }

  // C/D layout: lane = N col; VGPR q -> M = (lane>>4)*8 + q
  int mhigh = lane >> 4;
#pragma unroll
  for (int nt = 0; nt < 4; ++nt) {
    int ncol = bn + nt * 16 + l15;
    if (ncol >= N) continue;
#pragma unroll
    for (int q = 0; q < 8; ++q) {
      int mrow = bm + wave * 16 + mhigh * 8 + q;
      if (mrow < M) Cmat[(size_t)mrow * N + ncol] = acc[nt].f[q];
    }
  }
}

// ----------------------------------------------------------- GAT parts ----
__global__ void node_scores_kernel(const float* __restrict__ hf,
                                   const float* __restrict__ a_s,
                                   const float* __restrict__ a_d,
                                   float* __restrict__ ssrc, float* __restrict__ sdst,
                                   int H, int C) {
  int idx = blockIdx.x * blockDim.x + threadIdx.x;
  if (idx >= N_NODES * H) return;
  int n = idx / H, h = idx - n * H;
  const float* hp = hf + ((size_t)n * H + h) * C;
  const float* ap = a_s + (size_t)h * C;
  const float* dp = a_d + (size_t)h * C;
  float s1 = 0.f, s2 = 0.f;
  for (int c = 0; c < C; ++c) {
    float v = hp[c];
    s1 += v * ap[c];
    s2 += v * dp[c];
  }
  ssrc[idx] = s1;
  sdst[idx] = s2;
}

__global__ void fill_kernel(float* __restrict__ p, float v, long long n) {
  long long i = (long long)blockIdx.x * blockDim.x + threadIdx.x;
  if (i < n) p[i] = v;
}

__global__ void edge_score_max_kernel(const long long* __restrict__ ei,
                                      const float* __restrict__ ssrc,
                                      const float* __restrict__ sdst,
                                      float* __restrict__ ebuf,
                                      float* __restrict__ mmax, int H) {
  int idx = blockIdx.x * blockDim.x + threadIdx.x;
  if (idx >= E_TOT * H) return;
  int e = idx / H, h = idx - e * H;
  int s, d;
  edge_sd(ei, e, s, d);
  float v = ssrc[s * H + h] + sdst[d * H + h];
  v = v > 0.f ? v : 0.2f * v;   // leaky_relu(0.2)
  ebuf[idx] = v;
  atomicMaxF(&mmax[d * H + h], v);
}

__global__ void edge_exp_sum_kernel(const long long* __restrict__ ei,
                                    const float* __restrict__ ebuf,
                                    const float* __restrict__ mmax,
                                    float* __restrict__ pbuf,
                                    float* __restrict__ zsum, int H) {
  int idx = blockIdx.x * blockDim.x + threadIdx.x;
  if (idx >= E_TOT * H) return;
  int e = idx / H, h = idx - e * H;
  int s, d;
  edge_sd(ei, e, s, d);
  float p = expf(ebuf[idx] - mmax[d * H + h]);
  pbuf[idx] = p;
  atomicAdd(&zsum[d * H + h], p);
}

__global__ void edge_alpha_kernel(const long long* __restrict__ ei,
                                  float* __restrict__ pbuf,
                                  const float* __restrict__ zsum, int H) {
  int idx = blockIdx.x * blockDim.x + threadIdx.x;
  if (idx >= E_TOT * H) return;
  int e = idx / H, h = idx - e * H;
  int s, d;
  edge_sd(ei, e, s, d);
  pbuf[idx] = pbuf[idx] / (zsum[d * H + h] + 1e-16f);
}

// one block per edge; alphas staged in LDS; C is a power of two -> shift
__global__ void aggregate_kernel(const long long* __restrict__ ei,
                                 const float* __restrict__ hf,
                                 const float* __restrict__ alpha,
                                 float* __restrict__ agg, int H, int cshift) {
  __shared__ float sal[16];
  int e = blockIdx.x;
  int s, d;
  edge_sd(ei, e, s, d);
  if (threadIdx.x < H) sal[threadIdx.x] = alpha[e * H + threadIdx.x];
  __syncthreads();
  int HC = H << cshift;
  const float* src = hf + (size_t)s * HC;
  float* dst = agg + (size_t)d * HC;
  for (int f = threadIdx.x; f < HC; f += blockDim.x) {
    atomicAdd(&dst[f], src[f] * sal[f >> cshift]);
  }
}

__global__ void head_mean_selu_fix_kernel(const float* __restrict__ agg,
                                          const float* __restrict__ bias,
                                          const float* __restrict__ x,
                                          float* __restrict__ out,
                                          int H, int C, int do_fix) {
  int idx = blockIdx.x * blockDim.x + threadIdx.x;
  if (idx >= N_NODES * C) return;
  int n = idx / C, c = idx - n * C;
  float s = 0.f;
  for (int h = 0; h < H; ++h) s += agg[((size_t)n * H + h) * C + c];
  float v = selu_f(s / (float)H + bias[c]);
  if (do_fix) {
    float x0 = x[n * 10 + 0], x1 = x[n * 10 + 1];
    if (c == 0) v = (x0 == 0.f) ? 0.f : ((x0 == 1.f) ? 1.f : v);  // down->0, up->1
    if (c == 1) v = (x1 == 1.f) ? 1.f : ((x1 == 0.f) ? 0.f : v);  // right->1, left->0
  }
  out[idx] = v;
}

__global__ void final_out_kernel(const float* __restrict__ x,
                                 const float* __restrict__ res,
                                 float* __restrict__ out) {
  int n = blockIdx.x * blockDim.x + threadIdx.x;
  if (n >= N_NODES) return;
  float x0 = x[n * 10 + 0], x1 = x[n * 10 + 1];
  float v0 = x0 + res[n * 2 + 0];
  float v1 = x1 + res[n * 2 + 1];
  v0 = (x0 == 0.f) ? 0.f : ((x0 == 1.f) ? 1.f : v0);
  v1 = (x1 == 1.f) ? 1.f : ((x1 == 0.f) ? 0.f : v1);
  out[n * 2 + 0] = v0;
  out[n * 2 + 1] = v1;
}

// ---------------------------------------------------------------- host ----
extern "C" void kernel_launch(void* const* d_in, const int* in_sizes, int n_in,
                              void* d_out, int out_size, void* d_ws, size_t ws_size,
                              hipStream_t stream) {
  const float*     x  = (const float*)d_in[0];
  const long long* ei = (const long long*)d_in[1];   // int64 per reference
  const float*     cf = (const float*)d_in[2];
  const float* cw[4] = {(const float*)d_in[3], (const float*)d_in[5],
                        (const float*)d_in[7], (const float*)d_in[9]};
  const float* cb[4] = {(const float*)d_in[4], (const float*)d_in[6],
                        (const float*)d_in[8], (const float*)d_in[10]};
  const float *gw[5], *gas[5], *gad[5], *gb[5];
  for (int i = 0; i < 5; ++i) {
    gw[i]  = (const float*)d_in[11 + 4 * i];
    gas[i] = (const float*)d_in[12 + 4 * i];
    gad[i] = (const float*)d_in[13 + 4 * i];
    gb[i]  = (const float*)d_in[14 + 4 * i];
  }

  float* ws = (float*)d_ws;
  size_t off = 0;
  auto alloc = [&](size_t n) {
    float* p = ws + off;
    off += (n + 255) & ~(size_t)255;
    return p;
  };
  float* cnnA  = alloc((size_t)64 * 4096);
  float* cnnB  = alloc((size_t)64 * 4096);
  float* gfeat = alloc(64);
  float* hA    = alloc((size_t)N_NODES * 256);
  float* hB    = alloc((size_t)N_NODES * 256);
  float* hfeat = alloc((size_t)N_NODES * 2048);
  float* agg   = alloc((size_t)N_NODES * 2048);
  float* ssrc  = alloc((size_t)N_NODES * 16);
  float* sdst  = alloc((size_t)N_NODES * 16);
  float* mmax  = alloc((size_t)N_NODES * 16);
  float* zsum  = alloc((size_t)N_NODES * 16);
  float* ebuf  = alloc((size_t)E_TOT * 16);
  float* pbuf  = alloc((size_t)E_TOT * 16);
  (void)ws_size; (void)in_sizes; (void)n_in; (void)out_size;

  // ---- CNN: (1,4,64,64) -> gfeat(24)
  conv3x3_selu_kernel<<<dim3(16, 16), 256, 0, stream>>>(cf,   cw[0], cb[0], cnnA,  4, 16);
  conv3x3_selu_kernel<<<dim3(16, 32), 256, 0, stream>>>(cnnA, cw[1], cb[1], cnnB, 16, 32);
  conv3x3_selu_kernel<<<dim3(16, 64), 256, 0, stream>>>(cnnB, cw[2], cb[2], cnnA, 32, 64);
  conv3x3_selu_kernel<<<dim3(16, 24), 256, 0, stream>>>(cnnA, cw[3], cb[3], cnnB, 64, 24);
  avgpool_kernel<<<24, 256, 0, stream>>>(cnnB, gfeat);
  build_h0_kernel<<<(N_NODES * 34 + 255) / 256, 256, 0, stream>>>(x, gfeat, hA);

  // ---- 5 GAT layers
  const int Fin[5] = {34, 64, 128, 256, 128};
  const int Hn[5]  = {8, 16, 8, 8, 16};
  const int Cn[5]  = {64, 128, 256, 128, 2};
  const int Cs[5]  = {6, 7, 8, 7, 1};       // log2(C)
  float* hin = hA;
  float* hout = hB;
  for (int L = 0; L < 5; ++L) {
    int H = Hn[L], C = Cn[L], HC = H * C, K = Fin[L];

    gemm_wmma_kernel<<<dim3((HC + BN - 1) / BN, (N_NODES + BM - 1) / BM),
                       128, 0, stream>>>(hin, gw[L], hfeat, N_NODES, K, HC);

    node_scores_kernel<<<(N_NODES * H + 255) / 256, 256, 0, stream>>>(
        hfeat, gas[L], gad[L], ssrc, sdst, H, C);

    long long nh  = (long long)N_NODES * H;
    long long nhc = (long long)N_NODES * HC;
    fill_kernel<<<(unsigned)((nh + 255) / 256), 256, 0, stream>>>(mmax, -INFINITY, nh);
    fill_kernel<<<(unsigned)((nh + 255) / 256), 256, 0, stream>>>(zsum, 0.f, nh);
    fill_kernel<<<(unsigned)((nhc + 255) / 256), 256, 0, stream>>>(agg, 0.f, nhc);

    int eh = E_TOT * H;
    edge_score_max_kernel<<<(eh + 255) / 256, 256, 0, stream>>>(ei, ssrc, sdst, ebuf, mmax, H);
    edge_exp_sum_kernel<<<(eh + 255) / 256, 256, 0, stream>>>(ei, ebuf, mmax, pbuf, zsum, H);
    edge_alpha_kernel<<<(eh + 255) / 256, 256, 0, stream>>>(ei, pbuf, zsum, H);
    aggregate_kernel<<<E_TOT, 256, 0, stream>>>(ei, hfeat, pbuf, agg, H, Cs[L]);

    int do_fix = (L < 4) ? 1 : 0;
    head_mean_selu_fix_kernel<<<(N_NODES * C + 255) / 256, 256, 0, stream>>>(
        agg, gb[L], x, hout, H, C, do_fix);

    float* t = hin; hin = hout; hout = t;
  }

  // ---- residual + boundary fix -> d_out (N_NODES x 2 f32)
  final_out_kernel<<<(N_NODES + 255) / 256, 256, 0, stream>>>(x, hin, (float*)d_out);
}
